// BasisGATLayer_19894288515657
// MI455X (gfx1250) — matrix operand order repaired and checked by reading, not compiled, not run on hardware
//
#include <hip/hip_runtime.h>

#define B_    4
#define S_    256
#define D_    128
#define H_    8
#define HD_   16
#define DD_   64
#define T_    257
#define TPAD_ 288
#define NEG_INF_ (-1000000000.0f)

typedef __attribute__((ext_vector_type(16))) _Float16 v16h;
typedef __attribute__((ext_vector_type(8)))  _Float16 v8h;
typedef __attribute__((ext_vector_type(8)))  float    v8f;

// Workspace layout (float offsets)
#define OFF_Q   0                        // (B,H,T,HD)   f32
#define OFF_K   131584                   // (B,H,T,HD)   f32
#define OFF_V   263168                   // (B,H,HD,TPAD) f32, transposed + zero-padded
#define OFF_SQ  410624                   // (B,H,T)      f32
#define OFF_SK  418848                   // (B,H,T)      f32
#define OFF_G   427072                   // (B,4,T)      f32 edge score table
#define OFF_A16 431184                   // (B,H,T,TPAD) f16 padded attn copy

// ---------------- zero v padding ----------------
__global__ void zero_vpad_kernel(float* __restrict__ ws) {
    int n = B_ * H_ * HD_ * TPAD_;
    int i = blockIdx.x * blockDim.x + threadIdx.x;
    if (i < n) ws[OFF_V + i] = 0.0f;
}

// ---------------- QKV projection: y = x @ W.T + b  (WMMA) ----------------
// one wave = one 16(token) x 16(outdim) tile; blockIdx: x=row tile (B*17), y=head, z=matrix
__global__ void qkv_wmma_kernel(const float* __restrict__ nv,
                                const float* __restrict__ Wq, const float* __restrict__ bq,
                                const float* __restrict__ Wk, const float* __restrict__ bk,
                                const float* __restrict__ Wv, const float* __restrict__ bv,
                                float* __restrict__ ws) {
    const int lane   = threadIdx.x;
    const int lane16 = lane & 15;
    const int hl     = lane >> 4;
    const int m  = blockIdx.z;              // 0=q 1=k 2=v
    const int ct = blockIdx.y;              // head (HD==16 -> col tile == head)
    const int rt = blockIdx.x;              // 0..B*17-1
    const int b  = rt / 17;
    const int t0 = (rt % 17) * 16;

    const float* W    = (m == 0) ? Wq : (m == 1) ? Wk : Wv;
    const float* bias = (m == 0) ? bq : (m == 1) ? bk : bv;
    const int n   = ct * 16 + lane16;
    const int tok = t0 + lane16;            // A-fragment row for this lane

    __builtin_prefetch(&W[n * D_], 0, 1);   // global_prefetch_b8

    v8f c = {};
#pragma unroll
    for (int kc = 0; kc < 4; ++kc) {
        const int K0 = kc * 32;
        // ---- A: 16x32 f16. Lane's 16 elems = 8 floats at K0+hl*8 and 8 at K0+16+hl*8.
        float av[16];
        if (tok < T_) {
            const float* pa = &nv[(size_t)(b * T_ + tok) * D_ + K0 + hl * 8];
            *(float4*)&av[0]  = *(const float4*)(pa + 0);
            *(float4*)&av[4]  = *(const float4*)(pa + 4);
            *(float4*)&av[8]  = *(const float4*)(pa + 16);
            *(float4*)&av[12] = *(const float4*)(pa + 20);
        } else {
#pragma unroll
            for (int e = 0; e < 16; ++e) av[e] = 0.0f;
        }
        // ---- B: 32x16 f16. B[k][n] = W[n][k]; lane's 16 elems = 8 floats at K0+hl*8 and 8 at K0+16+hl*8.
        float bvv[16];
        {
            const float* pb = &W[(size_t)n * D_ + K0 + hl * 8];
            *(float4*)&bvv[0]  = *(const float4*)(pb + 0);
            *(float4*)&bvv[4]  = *(const float4*)(pb + 4);
            *(float4*)&bvv[8]  = *(const float4*)(pb + 16);
            *(float4*)&bvv[12] = *(const float4*)(pb + 20);
        }
        v16h a, bfrag;
#pragma unroll
        for (int e = 0; e < 16; ++e) {
            a[e]     = (_Float16)av[e];
            bfrag[e] = (_Float16)bvv[e];
        }
        c = __builtin_amdgcn_wmma_f32_16x16x32_f16(false, a, false, bfrag,
                                                   (short)0, c, false, false);
    }
    const float bn = bias[n];
#pragma unroll
    for (int r = 0; r < 8; ++r) {
        int t = t0 + r + hl * 8;            // D: M = r + 8*half, N = lane16
        if (t < T_) {
            float y = c[r] + bn;
            int h = ct, d = lane16;
            if (m == 0)      ws[OFF_Q + (((b * H_ + h) * T_ + t) * HD_ + d)] = y;
            else if (m == 1) ws[OFF_K + (((b * H_ + h) * T_ + t) * HD_ + d)] = y;
            else             ws[OFF_V + (((b * H_ + h) * HD_ + d) * TPAD_ + t)] = y; // transposed
        }
    }
}

// ---------------- per-token scalars sq, sk and edge table g ----------------
__global__ void scalars_kernel(const float* __restrict__ de,
                               const float* __restrict__ aw,
                               float* __restrict__ ws) {
    int tid = blockIdx.x * blockDim.x + threadIdx.x;
    if (tid >= B_ * H_ * T_) return;
    int t = tid % T_;
    int h = (tid / T_) % H_;
    int b = tid / (T_ * H_);

    const float* qp = &ws[OFF_Q + (size_t)tid * HD_];
    const float* kp = &ws[OFF_K + (size_t)tid * HD_];
    float sq = 0.f, sk = 0.f;
#pragma unroll
    for (int d = 0; d < HD_; ++d) {
        sq += qp[d] * aw[d];
        sk += kp[d] * aw[HD_ + d];
    }
    ws[OFF_SQ + tid] = sq;
    ws[OFF_SK + tid] = sk;

    if (h < 4) {                              // g[b][h][t] = de[b,t-1,h*16:] . ae
        float g = 0.f;
        if (t >= 1) {
            const float* dep = &de[((size_t)b * S_ + (t - 1)) * DD_ + h * HD_];
#pragma unroll
            for (int d = 0; d < HD_; ++d) g += dep[d] * aw[2 * HD_ + d];
        }
        ws[OFF_G + (b * 4 + h) * T_ + t] = g;
    }
}

// ---------------- logits + softmax: one wave per attention row, 8 waves/block ---------
__global__ void attn_softmax_kernel(float* __restrict__ ws,
                                    const float* __restrict__ ab_p,
                                    float* __restrict__ attn_out) {
    const int row = blockIdx.x * 8 + (threadIdx.x >> 5);   // (b*H + h)*T + i
    if (row >= B_ * H_ * T_) return;
    const int i = row % T_;
    const int h = (row / T_) % H_;
    const int b = row / (T_ * H_);
    const int lane = threadIdx.x & 31;

    const float ab  = ab_p[0];
    const float sqv = ws[OFF_SQ + row];
    const float* skp = &ws[OFF_SK + (b * H_ + h) * T_];
    const float* g   = &ws[OFF_G + (b * 4) * T_];
    _Float16* a16row = ((_Float16*)(ws + OFF_A16)) + (size_t)row * TPAD_;

    const float rowE = (i >= 1 && h < 4) ? g[h * T_ + i] : 0.0f;

    float vals[9];
    float mx = -3.4e38f;
#pragma unroll
    for (int it = 0; it < 9; ++it) {
        int j = lane + it * 32;
        float L = -3.4e38f;
        if (j < T_) {
            bool masked = (j == 0) || (i == j);
            float se = 0.f;
            if (!masked) {
                if (i == 0)      se = g[(h & 3) * T_ + j];   // CLS row
                else if (h < 4)  se = rowE;                  // depends on i only
                else             se = g[(h - 4) * T_ + j];   // depends on j only
            }
            L = sqv + skp[j] + se + ab + (masked ? NEG_INF_ : 0.0f);
        }
        vals[it] = L;
        mx = fmaxf(mx, L);
    }
    for (int o = 16; o > 0; o >>= 1) mx = fmaxf(mx, __shfl_xor(mx, o, 32));
    float sum = 0.f;
#pragma unroll
    for (int it = 0; it < 9; ++it) {
        int j = lane + it * 32;
        float p = (j < T_) ? __expf(vals[it] - mx) : 0.0f;
        vals[it] = p;
        sum += p;
    }
    for (int o = 16; o > 0; o >>= 1) sum += __shfl_xor(sum, o, 32);
    const float inv = 1.0f / sum;
#pragma unroll
    for (int it = 0; it < 9; ++it) {
        int j = lane + it * 32;                       // 9*32 == TPAD_: covers padding too
        float p = vals[it] * inv;
        if (j < T_) {
            attn_out[(size_t)row * T_ + j] = p;       // exact f32 output
            a16row[j] = (_Float16)p;                  // padded f16 copy for WMMA
        } else {
            a16row[j] = (_Float16)0.0f;               // zero K-padding
        }
    }
}

// ---------------- ctx = attn @ v  (WMMA, K padded to 288) ----------------
__global__ void ctx_wmma_kernel(const float* __restrict__ ws,
                                float* __restrict__ out) {
    const int lane   = threadIdx.x;
    const int lane16 = lane & 15;
    const int hl     = lane >> 4;
    const int i0 = blockIdx.x * 16;           // 17 row tiles
    const int h  = blockIdx.y;
    const int b  = blockIdx.z;

    const _Float16* a16 = ((const _Float16*)(ws + OFF_A16))
                        + (size_t)((b * H_ + h) * T_) * TPAD_;
    const float* vrow = &ws[OFF_V + (size_t)((b * H_ + h) * HD_ + lane16) * TPAD_];
    const int i = i0 + lane16;                // A-fragment row for this lane

    v8f c = {};
#pragma unroll
    for (int kc = 0; kc < 9; ++kc) {
        const int J0 = kc * 32;
        // ---- A: padded f16 attn, two 16-byte runs per lane
        v8h g0, g1;
        if (i < T_) {
            const _Float16* pa = a16 + (size_t)i * TPAD_ + J0 + hl * 8;
            g0 = *(const v8h*)(pa + 0);
            g1 = *(const v8h*)(pa + 16);
        } else {
            g0 = (v8h)(_Float16)0.0f;
            g1 = (v8h)(_Float16)0.0f;
        }
        // ---- B: v transposed (d-major), two runs of 8 floats per lane
        float bvv[16];
        {
            const float* pb = vrow + J0 + hl * 8;
            *(float4*)&bvv[0]  = *(const float4*)(pb + 0);
            *(float4*)&bvv[4]  = *(const float4*)(pb + 4);
            *(float4*)&bvv[8]  = *(const float4*)(pb + 16);
            *(float4*)&bvv[12] = *(const float4*)(pb + 20);
        }
        v16h a, bfrag;
#pragma unroll
        for (int e = 0; e < 8; ++e) { a[e] = g0[e]; a[e + 8] = g1[e]; }
#pragma unroll
        for (int e = 0; e < 16; ++e) bfrag[e] = (_Float16)bvv[e];
        c = __builtin_amdgcn_wmma_f32_16x16x32_f16(false, a, false, bfrag,
                                                   (short)0, c, false, false);
    }
#pragma unroll
    for (int r = 0; r < 8; ++r) {
        int ii = i0 + r + hl * 8;
        if (ii < T_)
            out[(((size_t)b * T_ + ii) * H_ + h) * HD_ + lane16] = c[r];
    }
}

extern "C" void kernel_launch(void* const* d_in, const int* in_sizes, int n_in,
                              void* d_out, int out_size, void* d_ws, size_t ws_size,
                              hipStream_t stream) {
    const float* de = (const float*)d_in[0];
    const float* nv = (const float*)d_in[1];
    const float* Wq = (const float*)d_in[2];
    const float* bq = (const float*)d_in[3];
    const float* Wk = (const float*)d_in[4];
    const float* bk = (const float*)d_in[5];
    const float* Wv = (const float*)d_in[6];
    const float* bv = (const float*)d_in[7];
    const float* aw = (const float*)d_in[8];
    const float* ab = (const float*)d_in[9];

    float* ws   = (float*)d_ws;
    float* out  = (float*)d_out;                 // basis_outputs (B,T,H,HD)
    float* attn = out + B_ * T_ * H_ * HD_;      // attn (B,H,T,T)

    {
        int n = B_ * H_ * HD_ * TPAD_;
        zero_vpad_kernel<<<(n + 255) / 256, 256, 0, stream>>>(ws);
    }
    qkv_wmma_kernel<<<dim3(B_ * 17, H_, 3), 32, 0, stream>>>(nv, Wq, bq, Wk, bk, Wv, bv, ws);
    {
        int n = B_ * H_ * T_;
        scalars_kernel<<<(n + 255) / 256, 256, 0, stream>>>(de, aw, ws);
    }
    {
        int rows = B_ * H_ * T_;
        attn_softmax_kernel<<<(rows + 7) / 8, 256, 0, stream>>>(ws, ab, attn);
    }
    ctx_wmma_kernel<<<dim3(17, H_, B_), 32, 0, stream>>>(ws, out);
}
